// MultiheadAttention_18451179504495
// MI455X (gfx1250) — matrix operand order repaired and checked by reading, not compiled, not run on hardware
//
#include <hip/hip_runtime.h>
#include <hip/hip_bf16.h>

// MHA forward for MI455X (gfx1250): bf16 WMMA (16x16x32, fp32 accum).
// B=4, S=2048, D=1024, H=16, HD=64.
// Round 4: GEMM waves now compute 32x64 tiles (8 accumulators) so every B
// fragment feeds 2 WMMAs and every A fragment feeds 4 — more matrix ops per
// loaded byte and 8 independent WMMAs to hide each fragment-load latency.

typedef __bf16 bf16;
typedef __attribute__((ext_vector_type(16))) bf16  v16bf;
typedef __attribute__((ext_vector_type(8)))  bf16  v8bf;
typedef __attribute__((ext_vector_type(8)))  float v8f;

#define WMMA_BF16(A, Bm, C) \
  __builtin_amdgcn_wmma_f32_16x16x32_bf16(false, (A), false, (Bm), (short)0, (C), false, false)

static constexpr int Bc = 4, Sc = 2048, Dc = 1024, Hc = 16, HDc = 64;
static constexpr size_t QKV_ELEMS = (size_t)Bc * Hc * Sc * HDc;  // 8388608
static constexpr size_t W_ELEMS   = (size_t)Dc * Dc;             // 1048576

// ---- helpers ---------------------------------------------------------------

__device__ __forceinline__ void cvt8(const float* __restrict__ p, v16bf& d, int off) {
  float4 f0 = ((const float4*)p)[0];
  float4 f1 = ((const float4*)p)[1];
  d[off + 0] = (bf16)f0.x; d[off + 1] = (bf16)f0.y;
  d[off + 2] = (bf16)f0.z; d[off + 3] = (bf16)f0.w;
  d[off + 4] = (bf16)f1.x; d[off + 5] = (bf16)f1.y;
  d[off + 6] = (bf16)f1.z; d[off + 7] = (bf16)f1.w;
}

__device__ __forceinline__ void ldb8(const bf16* __restrict__ p, v16bf& d, int off) {
  v8bf t = *(const v8bf*)p;
#pragma unroll
  for (int i = 0; i < 8; ++i) d[off + i] = t[i];
}

__device__ __forceinline__ v16bf ldb16(const bf16* __restrict__ p) {
  v16bf b;
  ldb8(p, b, 0);
  ldb8(p + 8, b, 8);
  return b;
}

// ---- kernel 0: fp32 -> bf16 bulk convert (8 elems/thread) ------------------
__global__ __launch_bounds__(256) void to_bf16(const float* __restrict__ src,
                                               bf16* __restrict__ dst, int n8) {
  int i = blockIdx.x * blockDim.x + threadIdx.x;
  if (i >= n8) return;
  const float* p = src + (size_t)i * 8;
  float4 f0 = ((const float4*)p)[0];
  float4 f1 = ((const float4*)p)[1];
  v8bf o;
  o[0] = (bf16)f0.x; o[1] = (bf16)f0.y; o[2] = (bf16)f0.z; o[3] = (bf16)f0.w;
  o[4] = (bf16)f1.x; o[5] = (bf16)f1.y; o[6] = (bf16)f1.z; o[7] = (bf16)f1.w;
  *(v8bf*)(dst + (size_t)i * 8) = o;
}

// ---- shared GEMM core: Y[m,n] = sum_k A[m,k] * W[n,k], 32x64 tile/wave -----
// Double-buffered fragment pipeline; acc[2][4] = 8 independent WMMAs/step.
template <typename StoreFn>
__device__ __forceinline__ void gemm_32x64(const bf16* __restrict__ ap0,
                                           const bf16* __restrict__ ap1,
                                           const bf16* const (&bp)[4],
                                           int half, StoreFn&& store) {
  v8f acc[2][4] = {{v8f{}, v8f{}, v8f{}, v8f{}}, {v8f{}, v8f{}, v8f{}, v8f{}}};

  v16bf aA[2], bA[4], aB[2], bB[4];
  auto load_frags = [&](int kb, v16bf (&a)[2], v16bf (&bb)[4]) {
    a[0] = ldb16(ap0 + kb + half * 16);
    a[1] = ldb16(ap1 + kb + half * 16);
#pragma unroll
    for (int j = 0; j < 4; ++j) bb[j] = ldb16(bp[j] + kb + half * 16);
  };
  auto do_wmma = [&](const v16bf (&a)[2], const v16bf (&bb)[4]) {
#pragma unroll
    for (int j = 0; j < 4; ++j) {
      acc[0][j] = WMMA_BF16(a[0], bb[j], acc[0][j]);
      acc[1][j] = WMMA_BF16(a[1], bb[j], acc[1][j]);
    }
  };

  load_frags(0, aA, bA);
#pragma unroll 2
  for (int kb = 0; kb < Dc; kb += 64) {
    if (kb + 128 < Dc) {
      __builtin_prefetch(ap0 + kb + 128, 0, 1);
      __builtin_prefetch(ap1 + kb + 128, 0, 1);
      __builtin_prefetch(bp[0] + kb + 128, 0, 1);
      __builtin_prefetch(bp[2] + kb + 128, 0, 1);
    }
    load_frags(kb + 32, aB, bB);      // in-flight while buffer A computes
    do_wmma(aA, bA);
    if (kb + 64 < Dc) load_frags(kb + 64, aA, bA);
    do_wmma(aB, bB);
  }
  store(acc);
}

// ---- kernel 1: fused Q/K/V projection (which = blockIdx.z) -----------------
__global__ __launch_bounds__(256) void proj_qkv(
    const bf16* __restrict__ X,
    const bf16* __restrict__ Wq, const bf16* __restrict__ Wk,
    const bf16* __restrict__ Wv,
    bf16* __restrict__ Qo, bf16* __restrict__ Ko, bf16* __restrict__ Vt) {
  const int wave = threadIdx.x >> 5;
  const int lane = threadIdx.x & 31;
  const int half = lane >> 4;
  const int lm   = lane & 15;
  const int which = blockIdx.z;

  const int widx = blockIdx.x * 8 + wave;   // 0..4095
  const int m0   = (widx >> 4) * 32;        // 32-row tile base
  const int n0   = (widx & 15) * 64;        // 64-wide col strip

  const bf16* W = (which == 0) ? Wq : (which == 1) ? Wk : Wv;
  const bf16* ap0 = X + (size_t)(m0 + lm) * Dc;
  const bf16* ap1 = X + (size_t)(m0 + 16 + lm) * Dc;
  const bf16* bp[4];
#pragma unroll
  for (int j = 0; j < 4; ++j) bp[j] = W + (size_t)(n0 + j * 16 + lm) * Dc;

  gemm_32x64(ap0, ap1, bp, half, [&](const v8f (&acc)[2][4]) {
#pragma unroll
    for (int mi = 0; mi < 2; ++mi) {
#pragma unroll
      for (int j = 0; j < 4; ++j) {
#pragma unroll
        for (int r = 0; r < 8; ++r) {
          const int row = m0 + mi * 16 + r + half * 8;
          const int col = n0 + j * 16 + lm;
          const int b = row >> 11, s = row & (Sc - 1);
          const int h = col >> 6,  hd = col & (HDc - 1);
          const bf16 v = (bf16)acc[mi][j][r];
          if (which == 0)      Qo[((size_t)(b * Hc + h) * Sc + s) * HDc + hd] = v;
          else if (which == 1) Ko[((size_t)(b * Hc + h) * Sc + s) * HDc + hd] = v;
          else                 Vt[((size_t)(b * Hc + h) * HDc + hd) * Sc + s] = v;
        }
      }
    }
  });
}

// ---- kernel 2: causal attention, one wave per (b,h,q-tile) -----------------
// Plain-exp softmax (no max subtraction) faithful to reference; -1e9 mask ==
// exp->0, so masked tiles are skipped / zeroed analytically.
__global__ __launch_bounds__(256) void attn_fwd(
    const bf16* __restrict__ Q, const bf16* __restrict__ K,
    const bf16* __restrict__ Vt, bf16* __restrict__ O) {
  __shared__ float plds[8][16 * 32];   // per-wave P staging (C-layout -> A-layout)

  const int wave = threadIdx.x >> 5;
  const int lane = threadIdx.x & 31;
  const int half = lane >> 4;
  const int lm   = lane & 15;

  const int bh    = blockIdx.y;
  const int qtile = blockIdx.x * 8 + wave;
  const int qbase = qtile * 16;

  const bf16* Qb = Q  + (size_t)bh * Sc * HDc;
  const bf16* Kb = K  + (size_t)bh * Sc * HDc;
  const bf16* Vb = Vt + (size_t)bh * HDc * Sc;

  v16bf qf[2];
#pragma unroll
  for (int c = 0; c < 2; ++c) {
    const bf16* qp = Qb + (size_t)(qbase + lm) * HDc + c * 32;
    qf[c] = ldb16(qp + half * 16);
  }

  v16bf ones;
#pragma unroll
  for (int i = 0; i < 16; ++i) ones[i] = (bf16)1.0f;

  v8f o[4] = {v8f{}, v8f{}, v8f{}, v8f{}};
  v8f den  = v8f{};
  float* myp = plds[wave];

  // K-fragment ping-pong buffers: [t = key half-tile][c = d chunk]
  v16bf kfA[2][2], kfB[2][2];
  auto load_k = [&](int kb, v16bf (&kf)[2][2]) {
#pragma unroll
    for (int t = 0; t < 2; ++t)
#pragma unroll
      for (int c = 0; c < 2; ++c)
        kf[t][c] = ldb16(Kb + (size_t)(kb + t * 16 + lm) * HDc + c * 32 + half * 16);
  };

  // One 32-wide key tile: scores -> exp/mask -> LDS relayout -> P@V, P@1.
  auto body = [&](int kb, const v16bf (&kf)[2][2]) {
    // V fragments first: latency hidden behind score WMMAs + exp + LDS.
    v16bf vf[4];
#pragma unroll
    for (int j = 0; j < 4; ++j)
      vf[j] = ldb16(Vb + (size_t)(j * 16 + lm) * Sc + kb + half * 16);

#pragma unroll
    for (int t = 0; t < 2; ++t) {
      v8f s = v8f{};
      s = WMMA_BF16(qf[0], kf[t][0], s);
      s = WMMA_BF16(qf[1], kf[t][1], s);
#pragma unroll
      for (int r = 0; r < 8; ++r) {
        const int grow = qbase + r + half * 8;
        const int gcol = kb + t * 16 + lm;
        const float e = (gcol <= grow) ? __expf(s[r] * 0.125f) : 0.0f;
        myp[(r + half * 8) * 32 + t * 16 + lm] = e;
      }
    }

    v16bf p;
    cvt8(myp + lm * 32 + half * 16, p, 0);
    cvt8(myp + lm * 32 + half * 16 + 8, p, 8);

#pragma unroll
    for (int j = 0; j < 4; ++j) o[j] = WMMA_BF16(p, vf[j], o[j]);
    den = WMMA_BF16(p, ones, den);
  };

  const int ntiles = (qbase + 47) >> 5;  // causal coverage of k <= q
  load_k(0, kfA);
  for (int kt = 0; kt < ntiles; kt += 2) {
    if (kt + 1 < ntiles) load_k((kt + 1) * 32, kfB);  // overlap with tile kt
    body(kt * 32, kfA);
    if (kt + 1 >= ntiles) break;
    if (kt + 2 < ntiles) load_k((kt + 2) * 32, kfA);  // overlap with tile kt+1
    body((kt + 1) * 32, kfB);
  }

  const int b = bh >> 4, h = bh & 15;
#pragma unroll
  for (int j = 0; j < 4; ++j) {
#pragma unroll
    for (int r = 0; r < 8; ++r) {
      const float val = o[j][r] / (den[r] + 1e-10f);
      const int grow = qbase + r + half * 8;
      O[((size_t)(b * Sc + grow)) * Dc + h * HDc + j * 16 + lm] = (bf16)val;
    }
  }
}

// ---- kernel 3: output projection, fp32 result ------------------------------
__global__ __launch_bounds__(256) void out_proj(
    const bf16* __restrict__ O, const bf16* __restrict__ Wo,
    float* __restrict__ Y) {
  const int wave = threadIdx.x >> 5;
  const int lane = threadIdx.x & 31;
  const int half = lane >> 4;
  const int lm   = lane & 15;

  const int widx = blockIdx.x * 8 + wave;   // 0..4095
  const int m0   = (widx >> 4) * 32;
  const int n0   = (widx & 15) * 64;

  const bf16* ap0 = O + (size_t)(m0 + lm) * Dc;
  const bf16* ap1 = O + (size_t)(m0 + 16 + lm) * Dc;
  const bf16* bp[4];
#pragma unroll
  for (int j = 0; j < 4; ++j) bp[j] = Wo + (size_t)(n0 + j * 16 + lm) * Dc;

  gemm_32x64(ap0, ap1, bp, half, [&](const v8f (&acc)[2][4]) {
#pragma unroll
    for (int mi = 0; mi < 2; ++mi) {
#pragma unroll
      for (int j = 0; j < 4; ++j) {
#pragma unroll
        for (int r = 0; r < 8; ++r) {
          Y[(size_t)(m0 + mi * 16 + r + half * 8) * Dc + n0 + j * 16 + lm] =
              acc[mi][j][r];
        }
      }
    }
  });
}

// ---- launch ----------------------------------------------------------------
extern "C" void kernel_launch(void* const* d_in, const int* in_sizes, int n_in,
                              void* d_out, int out_size, void* d_ws, size_t ws_size,
                              hipStream_t stream) {
  const float* x  = (const float*)d_in[0];
  // d_in[1] = mask: causal -1e9 handled analytically (exp(-1e9) == 0)
  const float* Wq = (const float*)d_in[2];
  const float* Wk = (const float*)d_in[3];
  const float* Wv = (const float*)d_in[4];
  const float* Wo = (const float*)d_in[5];

  bf16* base = (bf16*)d_ws;
  bf16* Qw  = base;                        // [B,H,S,HD]
  bf16* Kw  = Qw + QKV_ELEMS;              // [B,H,S,HD]
  bf16* Vt  = Kw + QKV_ELEMS;              // [B,H,HD,S] (transposed)
  bf16* Ow  = Vt + QKV_ELEMS;              // [B,S,D]
  bf16* Xb  = Ow + QKV_ELEMS;              // [B*S, D] bf16 copy of x
  bf16* Wqb = Xb + QKV_ELEMS;
  bf16* Wkb = Wqb + W_ELEMS;
  bf16* Wvb = Wkb + W_ELEMS;
  bf16* Wob = Wvb + W_ELEMS;
  float* Y  = (float*)d_out;

  dim3 blk(256);
  to_bf16<<<dim3((int)(QKV_ELEMS / 8 / 256)), blk, 0, stream>>>(x,  Xb,  (int)(QKV_ELEMS / 8));
  to_bf16<<<dim3((int)(W_ELEMS / 8 / 256)),   blk, 0, stream>>>(Wq, Wqb, (int)(W_ELEMS / 8));
  to_bf16<<<dim3((int)(W_ELEMS / 8 / 256)),   blk, 0, stream>>>(Wk, Wkb, (int)(W_ELEMS / 8));
  to_bf16<<<dim3((int)(W_ELEMS / 8 / 256)),   blk, 0, stream>>>(Wv, Wvb, (int)(W_ELEMS / 8));
  to_bf16<<<dim3((int)(W_ELEMS / 8 / 256)),   blk, 0, stream>>>(Wo, Wob, (int)(W_ELEMS / 8));

  proj_qkv<<<dim3(512, 1, 3), blk, 0, stream>>>(Xb, Wqb, Wkb, Wvb, Qw, Kw, Vt);
  attn_fwd<<<dim3(16, 64), blk, 0, stream>>>(Qw, Kw, Vt, Ow);
  out_proj<<<dim3(512), blk, 0, stream>>>(Ow, Wob, Y);
}
